// SelfAttention_17317308137994
// MI455X (gfx1250) — matrix operand order repaired and checked by reading, not compiled, not run on hardware
//
#include <hip/hip_runtime.h>

typedef __attribute__((ext_vector_type(16))) _Float16 v16h;
typedef __attribute__((ext_vector_type(8)))  _Float16 v8h;
typedef __attribute__((ext_vector_type(4)))  _Float16 v4h;
typedef __attribute__((ext_vector_type(8)))  float    v8f;
typedef __attribute__((ext_vector_type(4)))  float    v4f;

#define BATCH  4096
#define LSEQ   50
#define LPAD   64
#define FDIM   128
#define HHEADS 8
#define DQKD   128
#define DVD    16

// LDS region offsets (bytes), all 32-aligned. Total 64512 <= 64KB.
#define OFF_X   0        // x f16, rows 0..49 valid ([.][128]); padded-row reads spill into stage (harmless)
#define OFF_ST  12800    // W stage f16 64x128 = 16384; reused as f32 scores 64x64
#define OFF_Q   29184    // q f16 64x128 = 16384
#define OFF_K   45568    // k f16 64x128 = 16384
#define OFF_VT  61952    // v^T f16 16x64 = 2048
#define OFF_RED 64000    // softmax reductions: 64 f32 max + 64 f32 inv = 512
#define SMEM_BYTES 64512

// ---- WMMA fragment helpers (wave32, contiguous k = 16*(lane/16) + s, same for A and B) ----

// A-matrix 16x32 f16: lanes 0-15 -> row M=lane, K=0..15;
// lanes 16-31 -> row M=lane-16, K=16..31
__device__ __forceinline__ v16h load_a_frag(const _Float16* p, int ld, int row0, int k0) {
  const int lane = threadIdx.x & 31;
  const int r  = row0 + (lane & 15);
  const int kb = k0 + ((lane & 16) ? 16 : 0);
  return *(const v16h*)(p + r * ld + kb);
}

// B-matrix 32x16 f16: lanes 0-15 -> col N=lane, K=0..15; lanes 16-31 -> col N=lane-16, K=16..31.
// Source stored [n][k] (K contiguous per row).
__device__ __forceinline__ v16h load_b_frag(const _Float16* p, int ld, int n0, int k0) {
  const int lane = threadIdx.x & 31;
  const int n  = n0 + (lane & 15);
  const int kb = k0 + ((lane & 16) ? 16 : 0);
  return *(const v16h*)(p + n * ld + kb);
}

__device__ __forceinline__ v8f wmma_f16(v16h a, v16h b, v8f c) {
  return __builtin_amdgcn_wmma_f32_16x16x32_f16(false, a, false, b, (short)0, c, false, false);
}

// One projection pass: stage 64 transposed columns of W (float4 reads), compute
// two 16x16 tiles per wave (A fragments loaded once, reused for both N tiles).
__device__ __forceinline__ void proj_pass(
    const _Float16* s_x, _Float16* s_st, _Float16* dst,
    const float* __restrict__ W, const float* __restrict__ bias,
    int h, int half, int tid, int wave, int col, int rofs)
{
  __syncthreads();   // previous consumers of s_st done
  for (int i = tid; i < (64 * FDIM) / 4; i += 256) {
    const int k  = i >> 4;             // 0..127
    const int n4 = (i & 15) * 4;       // 0..60, consecutive tid -> consecutive n4 (coalesced b128)
    const v4f wv = *(const v4f*)(W + (size_t)h * FDIM * DQKD + (size_t)k * DQKD + 64 * half + n4);
#pragma unroll
    for (int j = 0; j < 4; ++j)
      s_st[(n4 + j) * FDIM + k] = (_Float16)wv[j];
  }
  __syncthreads();

  const int m0  = (wave & 3) * 16;
  const int nl0 = (wave >> 2) * 32;
  v16h a[4];
#pragma unroll
  for (int kk = 0; kk < 4; ++kk) a[kk] = load_a_frag(s_x, FDIM, m0, kk * 32);
#pragma unroll
  for (int nt = 0; nt < 2; ++nt) {
    const int nl = nl0 + nt * 16;
    v8f acc = {};
#pragma unroll
    for (int kk = 0; kk < 4; ++kk)
      acc = wmma_f16(a[kk], load_b_frag(s_st, FDIM, nl, kk * 32), acc);
    const int n0 = 64 * half + nl;
    const float bsc = bias[h * DQKD + n0 + col];
#pragma unroll
    for (int r = 0; r < 8; ++r)
      dst[(m0 + r + rofs) * DQKD + n0 + col] = (_Float16)(acc[r] + bsc);
  }
}

__global__ __launch_bounds__(256) void attn_bh_kernel(
    const float* __restrict__ hid,
    const float* __restrict__ Wq, const float* __restrict__ bq,
    const float* __restrict__ Wk, const float* __restrict__ bk,
    const float* __restrict__ Wv, const float* __restrict__ bv,
    float* __restrict__ out)
{
  __shared__ __align__(32) unsigned char smem[SMEM_BYTES];
  _Float16* s_x   = (_Float16*)(smem + OFF_X);
  _Float16* s_st  = (_Float16*)(smem + OFF_ST);
  _Float16* s_q   = (_Float16*)(smem + OFF_Q);
  _Float16* s_k   = (_Float16*)(smem + OFF_K);
  _Float16* s_vt  = (_Float16*)(smem + OFF_VT);
  float*    s_sc  = (float*)(smem + OFF_ST);      // reuse stage after QKV
  _Float16* s_att = (_Float16*)(smem + OFF_X);    // reuse x after QKV
  float*    s_mx  = (float*)(smem + OFF_RED);
  float*    s_inv = (float*)(smem + OFF_RED + 256);

  const int tid  = threadIdx.x;
  const int wave = tid >> 5;
  const int lane = tid & 31;
  const int b = blockIdx.x >> 3;
  const int h = blockIdx.x & 7;
  const int col  = lane & 15;
  const int rofs = (lane & 16) ? 8 : 0;

  // ---- Phase 0: stage x rows 0..49 (float4 reads, packed f16 b64 stores) ----
  for (int i = tid; i < (LSEQ * FDIM) / 4; i += 256) {
    const v4f xv = *(const v4f*)(hid + (size_t)b * (LSEQ * FDIM) + (size_t)i * 4);
    v4h hv;
#pragma unroll
    for (int j = 0; j < 4; ++j) hv[j] = (_Float16)xv[j];
    *(v4h*)(s_x + i * 4) = hv;
  }

  // ---- Phase 1: q, k (2 passes each), then v ----
  proj_pass(s_x, s_st, s_q, Wq, bq, h, 0, tid, wave, col, rofs);
  proj_pass(s_x, s_st, s_q, Wq, bq, h, 1, tid, wave, col, rofs);
  proj_pass(s_x, s_st, s_k, Wk, bk, h, 0, tid, wave, col, rofs);
  proj_pass(s_x, s_st, s_k, Wk, bk, h, 1, tid, wave, col, rofs);

  __syncthreads();
  for (int i = tid; i < (DVD * FDIM) / 4; i += 256) {
    const int k  = i >> 2;
    const int d4 = (i & 3) * 4;
    const v4f wv = *(const v4f*)(Wv + (size_t)h * FDIM * DVD + (size_t)k * DVD + d4);
#pragma unroll
    for (int j = 0; j < 4; ++j)
      s_st[(d4 + j) * FDIM + k] = (_Float16)wv[j];
  }
  __syncthreads();
  if (wave < 4) {
    const int m0 = wave * 16;
    v8f acc = {};
#pragma unroll
    for (int kk = 0; kk < 4; ++kk)
      acc = wmma_f16(load_a_frag(s_x, FDIM, m0, kk * 32),
                     load_b_frag(s_st, FDIM, 0, kk * 32), acc);
    const float bsc = bv[h * DVD + col];
    v8h pk;                       // contiguous in m -> one packed b128 store
#pragma unroll
    for (int r = 0; r < 8; ++r) {
      const int m = m0 + rofs + r;
      pk[r] = (_Float16)((m < LSEQ) ? (acc[r] + bsc) : 0.0f);
    }
    *(v8h*)(s_vt + col * LPAD + m0 + rofs) = pk;
  }
  __syncthreads();

  // ---- Phase 2: scores = q . k^T, leaky-relu, f32 into reused stage region ----
  {
    const int l0 = (wave & 3) * 16;
    const int mh = (wave >> 2) * 32;
    v16h a[4];
#pragma unroll
    for (int kk = 0; kk < 4; ++kk) a[kk] = load_a_frag(s_q, DQKD, l0, kk * 32);
#pragma unroll
    for (int mt = 0; mt < 2; ++mt) {
      const int m0 = mh + mt * 16;
      v8f acc = {};
#pragma unroll
      for (int kk = 0; kk < 4; ++kk)
        acc = wmma_f16(a[kk], load_b_frag(s_k, DQKD, m0, kk * 32), acc);
#pragma unroll
      for (int r = 0; r < 8; ++r) {
        float s = acc[r];
        s = (s > 0.0f) ? s : 0.1f * s;            // leaky_relu(0.1)
        s_sc[(l0 + r + rofs) * LPAD + m0 + col] = s;
      }
    }
  }
  __syncthreads();

  // ---- Phase 3a: per-column max and 1/sum over l = 0..49 ----
  if (tid < LPAD) {
    const int m = tid;
    float mx = 0.0f, inv = 0.0f;
    if (m < LSEQ) {
      mx = -1e30f;
      for (int l = 0; l < LSEQ; ++l) mx = fmaxf(mx, s_sc[l * LPAD + m]);
      float sum = 0.0f;
      for (int l = 0; l < LSEQ; ++l) sum += __expf(s_sc[l * LPAD + m] - mx);
      inv = 1.0f / sum;
    }
    s_mx[m] = mx; s_inv[m] = inv;
  }
  __syncthreads();

  // ---- Phase 3b: all 256 threads emit att row-major, packed b128 stores ----
  {
    const int l  = tid >> 2;            // 0..63
    const int mq = (tid & 3) * 16;      // 0,16,32,48
    v16h pk;
#pragma unroll
    for (int j = 0; j < 16; ++j) {
      const int m = mq + j;
      float val = 0.0f;
      if (l < LSEQ && m < LSEQ)
        val = __expf(s_sc[l * LPAD + m] - s_mx[m]) * s_inv[m];
      pk[j] = (_Float16)val;
    }
    *(v16h*)(s_att + l * LPAD + mq) = pk;
  }
  __syncthreads();

  // ---- Phase 4: out = att . v, relu, direct global store ----
  if (wave < 4) {
    const int l0 = wave * 16;
    v8f acc = {};
#pragma unroll
    for (int kk = 0; kk < 2; ++kk)
      acc = wmma_f16(load_a_frag(s_att, LPAD, l0, kk * 32),
                     load_b_frag(s_vt,  LPAD, 0,  kk * 32), acc);
#pragma unroll
    for (int r = 0; r < 8; ++r) {
      const int l = l0 + r + rofs;
      if (l < LSEQ)
        out[(size_t)b * (LSEQ * HHEADS * DVD) + l * (HHEADS * DVD) + h * DVD + col] =
            fmaxf(acc[r], 0.0f);
    }
  }
}

extern "C" void kernel_launch(void* const* d_in, const int* in_sizes, int n_in,
                              void* d_out, int out_size, void* d_ws, size_t ws_size,
                              hipStream_t stream) {
  const float* hid = (const float*)d_in[0];
  const float* Wq  = (const float*)d_in[1];
  const float* bq  = (const float*)d_in[2];
  const float* Wk  = (const float*)d_in[3];
  const float* bk  = (const float*)d_in[4];
  const float* Wv  = (const float*)d_in[5];
  const float* bv  = (const float*)d_in[6];
  float* out = (float*)d_out;
  (void)in_sizes; (void)n_in; (void)out_size; (void)d_ws; (void)ws_size;

  attn_bh_kernel<<<dim3(BATCH * HHEADS), dim3(256), 0, stream>>>(
      hid, Wq, bq, Wk, bk, Wv, bv, out);
}
